// MyLeNetRotateInvariantNew_nms_35845797052776
// MI455X (gfx1250) — matrix operand order repaired and checked by reading, not compiled, not run on hardware
//
#include <hip/hip_runtime.h>
#include <hip/hip_bf16.h>
#include <cmath>

#ifndef M_PI
#define M_PI 3.14159265358979323846
#endif

typedef __bf16 bf16;
typedef __attribute__((ext_vector_type(16))) __bf16 v16bf;
typedef __attribute__((ext_vector_type(8)))  __bf16 v8bf;
typedef __attribute__((ext_vector_type(8)))  float  v8f;

#define BN_EPS 1e-5f
#define NT 256

struct RotOffs { signed char dx[8][9]; signed char dy[8][9]; };

// ---------------------------------------------------------------------------
// Build rotated 3x3 kernels (scatter w taps to rounded rotated positions,
// accumulating collisions; non-separate layers cyclically roll input-channel
// blocks) and pack all conv weights into bf16 GEMM layout [CoutPad][KPad].
// ---------------------------------------------------------------------------
__global__ void build_weights_kernel(
    const float* __restrict__ w1, const float* __restrict__ w2,
    const float* __restrict__ w3, const float* __restrict__ w4,
    const float* __restrict__ w5, const float* __restrict__ w6,
    const float* __restrict__ w7,
    float* __restrict__ ker1, float* __restrict__ ker2, float* __restrict__ ker3,
    bf16* __restrict__ wg2, bf16* __restrict__ wg3, bf16* __restrict__ wg4,
    bf16* __restrict__ wg5, bf16* __restrict__ wg6, bf16* __restrict__ wg7,
    RotOffs ro)
{
  const int tid = threadIdx.x;
  for (int i = tid; i < 16*9;   i += NT) ker1[i] = 0.f;
  for (int i = tid; i < 16*144; i += NT) ker2[i] = 0.f;
  for (int i = tid; i < 32*144; i += NT) ker3[i] = 0.f;
  __syncthreads();

  // layer1: separate (no roll). out ch = rot*2+o, cin=1.
  for (int it = tid; it < 16; it += NT) {
    int rot = it >> 1, o = it & 1;
    for (int tap = 0; tap < 9; ++tap) {
      int lx = ro.dx[rot][tap], ly = ro.dy[rot][tap];
      ker1[(rot*2 + o)*9 + (1 - ly)*3 + (1 - lx)] += w1[o*9 + tap];
    }
  }
  // layer2: w2 [2,16,3,3], roll rotation blocks (cin/r = 2)
  for (int it = tid; it < 8*2*16; it += NT) {
    int rot = it >> 5, o = (it >> 4) & 1, ci = it & 15;
    int jblk = ci >> 1, c2 = ci & 1;
    int src = ((((jblk - rot) % 8) + 8) % 8) * 2 + c2;
    for (int tap = 0; tap < 9; ++tap) {
      int lx = ro.dx[rot][tap], ly = ro.dy[rot][tap];
      ker2[(((rot*2 + o)*16) + ci)*9 + (1 - ly)*3 + (1 - lx)] += w2[(o*16 + src)*9 + tap];
    }
  }
  // layer3: w3 [4,16,3,3]
  for (int it = tid; it < 8*4*16; it += NT) {
    int rot = it >> 6, o = (it >> 4) & 3, ci = it & 15;
    int jblk = ci >> 1, c2 = ci & 1;
    int src = ((((jblk - rot) % 8) + 8) % 8) * 2 + c2;
    for (int tap = 0; tap < 9; ++tap) {
      int lx = ro.dx[rot][tap], ly = ro.dy[rot][tap];
      ker3[(((rot*4 + o)*16) + ci)*9 + (1 - ly)*3 + (1 - lx)] += w3[(o*16 + src)*9 + tap];
    }
  }
  __syncthreads();

  // pack to bf16 GEMM layout (K padded with zeros to a multiple of 32)
  for (int i = tid; i < 16*160; i += NT) { int oc = i/160, k = i%160;
    wg2[i] = (bf16)(k < 144 ? ker2[oc*144 + k] : 0.f); }
  for (int i = tid; i < 32*160; i += NT) { int oc = i/160, k = i%160;
    wg3[i] = (bf16)(k < 144 ? ker3[oc*144 + k] : 0.f); }
  for (int i = tid; i < 32*288;  i += NT) wg4[i] = (bf16)w4[i];  // 288 = 32*9
  for (int i = tid; i < 64*288;  i += NT) wg5[i] = (bf16)w5[i];
  for (int i = tid; i < 64*576;  i += NT) wg6[i] = (bf16)w6[i];  // 576 = 64*9
  for (int i = tid; i < 16*1024; i += NT) { int oc = i >> 10;
    wg7[i] = (bf16)(oc < 10 ? w7[i] : 0.f); }                    // 1024 = 64*16
}

// ---------------------------------------------------------------------------
// Layer1 direct conv: cin=1, 3x3, pad 1, 16 rotated out channels. fp32 raw out.
// ---------------------------------------------------------------------------
__global__ void conv1_kernel(const float* __restrict__ x, const float* __restrict__ ker1,
                             float* __restrict__ out)
{
  __shared__ float sk[144];
  if (threadIdx.x < 144) sk[threadIdx.x] = ker1[threadIdx.x];
  __syncthreads();
  int idx = blockIdx.x*NT + threadIdx.x;
  if (idx >= 2048*784) return;
  int b = idx / 784, r = idx % 784, y = r / 28, xo = r % 28;
  float vin[9];
#pragma unroll
  for (int t = 0; t < 9; ++t) {
    int iy = y + t/3 - 1, ix = xo + t%3 - 1;
    vin[t] = ((unsigned)iy < 28u && (unsigned)ix < 28u) ? x[b*784 + iy*28 + ix] : 0.f;
  }
  size_t obase = (size_t)b*16*784 + (size_t)y*28 + xo;
#pragma unroll
  for (int oc = 0; oc < 16; ++oc) {
    float s = 0.f;
#pragma unroll
    for (int t = 0; t < 9; ++t) s += sk[oc*9 + t] * vin[t];
    out[obase + (size_t)oc*784] = s;
  }
}

// ---------------------------------------------------------------------------
// Implicit-GEMM conv on v_wmma_f32_16x16x32_bf16, fully compile-time geometry.
// One wave = 16 out-ch x 16 pixel tile; K = CIN*KS*KS zero-padded to KPAD.
// Weights pre-packed [CoutPad][KPAD] bf16 row-major; activations NCHW bf16.
// ---------------------------------------------------------------------------
template<int CIN, int H, int W, int KS, int PAD, int HOUT, int WOUT, int KPAD>
__device__ __forceinline__ float im2col_elem(const bf16* __restrict__ inb,
                                             int k, unsigned vym, unsigned vxm)
{
  constexpr int KK = KS * KS;
  if (k >= CIN * KK) return 0.f;              // compile-time-folded zero padding
  const int ci = k / KK, tap = k % KK;        // constants after unroll
  const int ty = tap / KS, tx = tap % KS;
  float v = 0.f;
  if (((vym >> ty) & 1u) && ((vxm >> tx) & 1u))
    v = (float)inb[ci * (H * W) + ty * W + tx];
  return v;
}

template<int CIN, int H, int W, int KS, int PAD, int HOUT, int WOUT, int KPAD>
__global__ void conv_wmma_kernel(const bf16* __restrict__ in, const bf16* __restrict__ wg,
                                 float* __restrict__ out, int B, int Cout)
{
  constexpr int HWo = HOUT * WOUT;
  const int lane   = threadIdx.x & 31;
  const int waveId = threadIdx.x >> 5;
  const int ptile  = blockIdx.x * (blockDim.x >> 5) + waveId;
  const int octile = blockIdx.y;
  const int lm = lane & 15;
  const int hi = lane >> 4;

  const int P = B * HWo;
  const int p = ptile*16 + lm;
  const bool pv = (p < P);
  int b = 0, oy = 0, ox = 0;
  if (pv) { b = p / HWo; int r = p % HWo; oy = r / WOUT; ox = r % WOUT; }

  // Per-lane validity bitmasks over the KSxKS tap window (pv folded in).
  unsigned vym = 0, vxm = 0;
#pragma unroll
  for (int t = 0; t < KS; ++t) {
    vym |= ((unsigned)(oy + t - PAD) < (unsigned)H ? 1u : 0u) << t;
    vxm |= ((unsigned)(ox + t - PAD) < (unsigned)W ? 1u : 0u) << t;
  }
  if (!pv) { vym = 0; vxm = 0; }

  const bf16* inb  = in + ((size_t)b * CIN * H * W + (size_t)(oy - PAD) * W + (ox - PAD));
  const bf16* wrow = wg + (size_t)(octile*16 + lm) * KPAD;

  v8f acc = {};
#pragma unroll
  for (int kc = 0; kc < KPAD; kc += 32) {
    // A fragment: contiguous K (lane-half h holds k = 16h..16h+15) -> two aligned 8xbf16 loads
    v8bf a0 = *(const v8bf*)(wrow + kc + hi*16);
    v8bf a1 = *(const v8bf*)(wrow + kc + hi*16 + 8);
    v16bf a;
#pragma unroll
    for (int e = 0; e < 8; ++e) { a[e] = a0[e]; a[e + 8] = a1[e]; }

    // B fragment: lane&15 = pixel column; lane-half selects K half.
    // Divergent halves keep every im2col index a compile-time constant.
    v16bf bm;
    if (hi == 0) {
#pragma unroll
      for (int e = 0; e < 16; ++e)
        bm[e] = (bf16)im2col_elem<CIN,H,W,KS,PAD,HOUT,WOUT,KPAD>(inb, kc + e, vym, vxm);
    } else {
#pragma unroll
      for (int e = 0; e < 16; ++e)
        bm[e] = (bf16)im2col_elem<CIN,H,W,KS,PAD,HOUT,WOUT,KPAD>(inb, kc + 16 + e, vym, vxm);
    }
    acc = __builtin_amdgcn_wmma_f32_16x16x32_bf16(false, a, false, bm,
                                                  (short)0, acc, false, false);
  }

  if (pv) {
#pragma unroll
    for (int j = 0; j < 8; ++j) {
      int oc = octile*16 + hi*8 + j;   // C/D layout: laneHi selects M 8..15
      if (oc < Cout)
        out[((size_t)(b*Cout + oc)*HOUT + oy)*WOUT + ox] = acc[j];
    }
  }
}

// ---------------------------------------------------------------------------
// Training-mode BN statistics: one block per channel, deterministic tree reduce.
// ---------------------------------------------------------------------------
__global__ void stats_kernel(const float* __restrict__ raw, float* __restrict__ mu,
                             float* __restrict__ rsig, int B, int C, int HW)
{
  const int c = blockIdx.x;
  float s = 0.f, s2 = 0.f;
  for (int b = 0; b < B; ++b) {
    const float* rp = raw + ((size_t)b*C + c)*HW;
    for (int sp = threadIdx.x; sp < HW; sp += NT) {
      float v = rp[sp];
      s += v; s2 += v*v;
    }
  }
  __shared__ float sh[NT], sh2[NT];
  sh[threadIdx.x] = s; sh2[threadIdx.x] = s2;
  __syncthreads();
  for (int st = NT/2; st > 0; st >>= 1) {
    if (threadIdx.x < st) { sh[threadIdx.x] += sh[threadIdx.x + st];
                            sh2[threadIdx.x] += sh2[threadIdx.x + st]; }
    __syncthreads();
  }
  if (threadIdx.x == 0) {
    float n = (float)B * (float)HW;
    float m = sh[0] / n;
    float var = sh2[0] / n - m*m;
    mu[c] = m; rsig[c] = rsqrtf(var + BN_EPS);
  }
}

// BN apply + ReLU -> bf16  (rot-BN == per-channel stats, gamma index = c % glen)
__global__ void bn_relu_kernel(const float* __restrict__ raw, bf16* __restrict__ outa,
                               const float* __restrict__ mu, const float* __restrict__ rsig,
                               const float* __restrict__ gamma, const float* __restrict__ beta,
                               int C, int HW, int glen, int total)
{
  int idx = blockIdx.x*NT + threadIdx.x;
  if (idx >= total) return;
  int c = (idx / HW) % C, gi = c % glen;
  float v = (raw[idx] - mu[c]) * rsig[c] * gamma[gi] + beta[gi];
  outa[idx] = (bf16)fmaxf(v, 0.f);
}

// BN apply + ReLU + 2x2 maxpool (stride 2, optional pad) -> bf16
__global__ void bn_relu_pool_kernel(const float* __restrict__ raw, bf16* __restrict__ outa,
                                    const float* __restrict__ mu, const float* __restrict__ rsig,
                                    const float* __restrict__ gamma, const float* __restrict__ beta,
                                    int B, int C, int H, int W, int Ho, int Wo, int ppad, int glen)
{
  int total = B * C * Ho * Wo;
  int idx = blockIdx.x*NT + threadIdx.x;
  if (idx >= total) return;
  int wo = idx % Wo; int t = idx / Wo;
  int ho = t % Ho;   t /= Ho;
  int c  = t % C;    int b = t / C;
  int gi = c % glen;
  float sc = rsig[c] * gamma[gi];
  float sb = beta[gi] - mu[c] * sc;
  float m = -1e30f;
#pragma unroll
  for (int dy = 0; dy < 2; ++dy)
#pragma unroll
    for (int dx = 0; dx < 2; ++dx) {
      int iy = ho*2 + dy - ppad, ix = wo*2 + dx - ppad;
      if ((unsigned)iy < (unsigned)H && (unsigned)ix < (unsigned)W) {
        float v = fmaxf(raw[((size_t)(b*C + c)*H + iy)*W + ix] * sc + sb, 0.f);
        m = fmaxf(m, v);
      }
    }
  outa[idx] = (bf16)m;
}

// ---------------------------------------------------------------------------
// nms: view [B, C/r, r, H, W] (rotation-minor: ch = c4*8 + rot), max over rot,
// mean of values strictly below the group max. Deterministic two-stage reduce.
// ---------------------------------------------------------------------------
__global__ void nms_partial_kernel(const bf16* __restrict__ act, float* __restrict__ part)
{
  const int total = 2048*4*196;
  float acc = 0.f;
  for (int i = blockIdx.x*NT + threadIdx.x; i < total; i += NT*256) {
    int sp = i % 196; int t = i / 196; int c4 = t % 4; int b = t / 4;
    size_t base = ((size_t)b*32 + c4*8)*196 + sp;
    float v[8]; float mx = -1e30f;
#pragma unroll
    for (int r2 = 0; r2 < 8; ++r2) { v[r2] = (float)act[base + (size_t)r2*196]; mx = fmaxf(mx, v[r2]); }
#pragma unroll
    for (int r2 = 0; r2 < 8; ++r2) if (v[r2] != mx) acc += v[r2];
  }
  __shared__ float sh[NT];
  sh[threadIdx.x] = acc;
  __syncthreads();
  for (int st = NT/2; st > 0; st >>= 1) {
    if (threadIdx.x < st) sh[threadIdx.x] += sh[threadIdx.x + st];
    __syncthreads();
  }
  if (threadIdx.x == 0) part[blockIdx.x] = sh[0];
}

__global__ void nms_final_kernel(const float* __restrict__ part, float* __restrict__ out_n)
{
  __shared__ float sh[NT];
  sh[threadIdx.x] = part[threadIdx.x];
  __syncthreads();
  for (int st = NT/2; st > 0; st >>= 1) {
    if (threadIdx.x < st) sh[threadIdx.x] += sh[threadIdx.x + st];
    __syncthreads();
  }
  if (threadIdx.x == 0) out_n[0] = sh[0] / (2048.f*32.f*196.f);
}

// Final: +bias then global max over 5x5 (max(x)+b == max(x+b))
__global__ void final_kernel(const float* __restrict__ raw7, const float* __restrict__ b7,
                             float* __restrict__ out)
{
  int idx = blockIdx.x*NT + threadIdx.x;
  if (idx >= 2048*10) return;
  int oc = idx % 10;
  const float* p = raw7 + (size_t)idx * 25;
  float m = p[0];
#pragma unroll
  for (int j = 1; j < 25; ++j) m = fmaxf(m, p[j]);
  out[idx] = m + b7[oc];
}

// ---------------------------------------------------------------------------
extern "C" void kernel_launch(void* const* d_in, const int* in_sizes, int n_in,
                              void* d_out, int out_size, void* d_ws, size_t ws_size,
                              hipStream_t stream) {
  const float* x  = (const float*)d_in[0];
  const float* w1 = (const float*)d_in[1];
  const float* w2 = (const float*)d_in[2];
  const float* w3 = (const float*)d_in[3];
  const float* w4 = (const float*)d_in[4];
  const float* w5 = (const float*)d_in[5];
  const float* w6 = (const float*)d_in[6];
  const float* w7 = (const float*)d_in[7];
  const float* g1 = (const float*)d_in[8];  const float* bb1 = (const float*)d_in[9];
  const float* g2 = (const float*)d_in[10]; const float* bb2 = (const float*)d_in[11];
  const float* g3 = (const float*)d_in[12]; const float* bb3 = (const float*)d_in[13];
  const float* g4 = (const float*)d_in[14]; const float* bb4 = (const float*)d_in[15];
  const float* g5 = (const float*)d_in[16]; const float* bb5 = (const float*)d_in[17];
  const float* g6 = (const float*)d_in[18]; const float* bb6 = (const float*)d_in[19];
  const float* b7 = (const float*)d_in[20];
  float* out = (float*)d_out;

  char* ws = (char*)d_ws;
  size_t off = 0;
  auto alloc = [&](size_t n) { off = (off + 255) & ~(size_t)255; size_t o = off; off += n; return o; };

  float* RAW  = (float*)(ws + alloc((size_t)2048*16*28*28*4)); // 102.8 MB raw conv out (reused)
  bf16*  A0   = (bf16*) (ws + alloc((size_t)2048*16*28*28*2)); // 51.4 MB bf16 activations (ping)
  bf16*  A1   = (bf16*) (ws + alloc((size_t)2048*16*14*14*2)); // 12.9 MB bf16 activations (pong)
  float* KER1 = (float*)(ws + alloc(16*9*4));
  float* KER2 = (float*)(ws + alloc(16*144*4));
  float* KER3 = (float*)(ws + alloc(32*144*4));
  bf16*  WG2  = (bf16*) (ws + alloc(16*160*2));
  bf16*  WG3  = (bf16*) (ws + alloc(32*160*2));
  bf16*  WG4  = (bf16*) (ws + alloc(32*288*2));
  bf16*  WG5  = (bf16*) (ws + alloc(64*288*2));
  bf16*  WG6  = (bf16*) (ws + alloc(64*576*2));
  bf16*  WG7  = (bf16*) (ws + alloc(16*1024*2));
  float* MU   = (float*)(ws + alloc(64*4));
  float* RSIG = (float*)(ws + alloc(64*4));
  float* PART = (float*)(ws + alloc(256*4));

  // Rotated tap offsets exactly as cal_rot_pad (np.around == round-half-even)
  RotOffs ro;
  for (int i = 0; i < 8; ++i) {
    double th = 2.0*M_PI/8.0*(double)i;
    double cs = cos(th), sn = sin(th);
    for (int tap = 0; tap < 9; ++tap) {
      double lx = (double)(tap % 3) - 1.0, ly = (double)(tap / 3) - 1.0;
      ro.dx[i][tap] = (signed char)std::nearbyint(lx*cs + ly*sn);
      ro.dy[i][tap] = (signed char)std::nearbyint(-lx*sn + ly*cs);
    }
  }

  build_weights_kernel<<<1, NT, 0, stream>>>(w1, w2, w3, w4, w5, w6, w7,
      KER1, KER2, KER3, WG2, WG3, WG4, WG5, WG6, WG7, ro);

  auto cdiv = [](int a, int b) { return (a + b - 1) / b; };
  auto wgrid = [&](int P) { return cdiv(cdiv(P, 16), 4); }; // 4 waves/block

  // L1: rot conv (direct) -> BN -> ReLU
  conv1_kernel<<<cdiv(2048*784, NT), NT, 0, stream>>>(x, KER1, RAW);
  stats_kernel<<<16, NT, 0, stream>>>(RAW, MU, RSIG, 2048, 16, 784);
  bn_relu_kernel<<<cdiv(2048*16*784, NT), NT, 0, stream>>>(RAW, A0, MU, RSIG, g1, bb1,
                                                           16, 784, 2, 2048*16*784);
  // L2: rot conv (WMMA) -> BN -> ReLU -> pool2
  conv_wmma_kernel<16,28,28,3,1,28,28,160><<<dim3(wgrid(2048*784), 1), 128, 0, stream>>>(
      A0, WG2, RAW, 2048, 16);
  stats_kernel<<<16, NT, 0, stream>>>(RAW, MU, RSIG, 2048, 16, 784);
  bn_relu_pool_kernel<<<cdiv(2048*16*196, NT), NT, 0, stream>>>(RAW, A1, MU, RSIG, g2, bb2,
      2048, 16, 28, 28, 14, 14, 0, 2);
  // L3: rot conv (WMMA) -> BN -> ReLU ; nms
  conv_wmma_kernel<16,14,14,3,1,14,14,160><<<dim3(wgrid(2048*196), 2), 128, 0, stream>>>(
      A1, WG3, RAW, 2048, 32);
  stats_kernel<<<32, NT, 0, stream>>>(RAW, MU, RSIG, 2048, 32, 196);
  bn_relu_kernel<<<cdiv(2048*32*196, NT), NT, 0, stream>>>(RAW, A0, MU, RSIG, g3, bb3,
                                                           32, 196, 4, 2048*32*196);
  nms_partial_kernel<<<256, NT, 0, stream>>>(A0, PART);
  nms_final_kernel<<<1, NT, 0, stream>>>(PART, out + 20480);
  // L4: conv (WMMA) -> BN -> ReLU -> pool2
  conv_wmma_kernel<32,14,14,3,1,14,14,288><<<dim3(wgrid(2048*196), 2), 128, 0, stream>>>(
      A0, WG4, RAW, 2048, 32);
  stats_kernel<<<32, NT, 0, stream>>>(RAW, MU, RSIG, 2048, 32, 196);
  bn_relu_pool_kernel<<<cdiv(2048*32*49, NT), NT, 0, stream>>>(RAW, A1, MU, RSIG, g4, bb4,
      2048, 32, 14, 14, 7, 7, 0, 32);
  // L5: conv (WMMA) -> BN -> ReLU
  conv_wmma_kernel<32,7,7,3,1,7,7,288><<<dim3(wgrid(2048*49), 4), 128, 0, stream>>>(
      A1, WG5, RAW, 2048, 64);
  stats_kernel<<<64, NT, 0, stream>>>(RAW, MU, RSIG, 2048, 64, 49);
  bn_relu_kernel<<<cdiv(2048*64*49, NT), NT, 0, stream>>>(RAW, A0, MU, RSIG, g5, bb5,
                                                          64, 49, 64, 2048*64*49);
  // L6: conv (WMMA) -> BN -> ReLU -> pool2(pad 1)
  conv_wmma_kernel<64,7,7,3,1,7,7,576><<<dim3(wgrid(2048*49), 4), 128, 0, stream>>>(
      A0, WG6, RAW, 2048, 64);
  stats_kernel<<<64, NT, 0, stream>>>(RAW, MU, RSIG, 2048, 64, 49);
  bn_relu_pool_kernel<<<cdiv(2048*64*16, NT), NT, 0, stream>>>(RAW, A1, MU, RSIG, g6, bb6,
      2048, 64, 7, 7, 4, 4, 1, 64);
  // L7: 4x4 conv pad 2 (WMMA, Cout 10 padded to 16) -> +bias, global max
  conv_wmma_kernel<64,4,4,4,2,5,5,1024><<<dim3(wgrid(2048*25), 1), 128, 0, stream>>>(
      A1, WG7, RAW, 2048, 10);
  final_kernel<<<cdiv(2048*10, NT), NT, 0, stream>>>(RAW, b7, out);
}
